// PointerNet_5265629905045
// MI455X (gfx1250) — matrix operand order repaired and checked by reading, not compiled, not run on hardware
//
#include <hip/hip_runtime.h>

// ---------------------------------------------------------------------------
// Pointer network for MI455X (gfx1250, wave32, WMMA).
// Persistent-kernel LSTM recurrence (grid sync via atomics), bf16 WMMA GEMMs
// with f32 accumulation. Recurrent weights staged once into LDS (bank-conflict
// padded) so the 512-step serial loop reads B-operands at WGP-local latency;
// activations/h stay in the (L2-resident) global workspace.
// ---------------------------------------------------------------------------

typedef __bf16 bf16_t;
typedef __bf16 v16bf __attribute__((ext_vector_type(16)));
typedef __bf16 v8bf  __attribute__((ext_vector_type(8)));
typedef float  v8f   __attribute__((ext_vector_type(8)));

static constexpr int B_ = 64, S_ = 512, E_ = 256, H_ = 512, T_ = 32;
static constexpr int G4 = 4 * H_;      // 2048 gate columns
static constexpr int KENC = E_ + H_;   // 768 concatenated K
static constexpr int LDW  = KENC + 8;  // padded LDS row (bank-conflict free)
static constexpr int LDWIN = H_ + 8;   // padded LDS row for W_in slice

static constexpr size_t ENC_LDS = (size_t)64 * LDW * sizeof(bf16_t);
static constexpr size_t DEC_LDS = (size_t)64 * LDW * sizeof(bf16_t)
                                + (size_t)16 * LDWIN * sizeof(bf16_t);

struct Ws {
  bf16_t *Wenc, *Wdec, *Win, *Wout, *Wctx;   // bf16 weights (K contiguous)
  bf16_t *Xenc, *Demb, *EncO;                // embedded inputs / encoder outputs
  bf16_t *Hbuf, *Ht, *Hid, *Decin;           // recurrent state (Hbuf: 2 slots)
  float  *Ctx, *Cbuf, *Inp, *Mask;
  unsigned *Sync;
};

// ----- WMMA fragment loaders (layouts per CDNA5 ISA 7.12.2) ------------------
__device__ __forceinline__ v16bf load_a16(const bf16_t* base, int ld, int row0,
                                          int k0, int lane) {
  // A 16x32 bf16: lane&15 = row, halves at k0+8*(lane>>4) and +16
  const bf16_t* p = base + (size_t)(row0 + (lane & 15)) * ld
                         + (k0 + ((lane >> 4) << 3));
  union { v16bf v; v8bf h[2]; } u;
  u.h[0] = *reinterpret_cast<const v8bf*>(p);
  u.h[1] = *reinterpret_cast<const v8bf*>(p + 16);
  return u.v;
}
__device__ __forceinline__ v16bf load_b16(const bf16_t* W, int ldk, int n0,
                                          int k0, int lane) {
  // B 32x16 bf16 from row-major W[N][K]: lane&15 = column n, K contiguous
  const bf16_t* p = W + (size_t)(n0 + (lane & 15)) * ldk
                      + (k0 + ((lane >> 4) << 3));
  union { v16bf v; v8bf h[2]; } u;
  u.h[0] = *reinterpret_cast<const v8bf*>(p);
  u.h[1] = *reinterpret_cast<const v8bf*>(p + 16);
  return u.v;
}
__device__ __forceinline__ v8f wmma_bf16(v16bf a, v16bf b, v8f c) {
  return __builtin_amdgcn_wmma_f32_16x16x32_bf16(false, a, false, b,
                                                 (short)0, c, false, false);
}
__device__ __forceinline__ float sigmoidf_(float x) {
  return 1.f / (1.f + __expf(-x));
}

// ----- device-scope grid barrier (persistent kernels) ------------------------
__device__ __forceinline__ void grid_sync(unsigned* cnt, unsigned target) {
  __threadfence();
  __syncthreads();
  if (threadIdx.x == 0) {
    atomicAdd(cnt, 1u);
    while (*((volatile unsigned*)cnt) < target) __builtin_amdgcn_s_sleep(2);
  }
  __syncthreads();
  __threadfence();
}

// ----- stage a [rows x ldk] weight slice into padded LDS ---------------------
__device__ __forceinline__ void stage_gate_slice(bf16_t* lds_w, const bf16_t* W,
                                                 int n0, int tid) {
  // rows: lr = gate*16 + nloc  ->  global row gate*H_ + n0 + nloc
  const int chunks_per_row = KENC / 8;                  // 96 x 16B
  for (int c = tid; c < 64 * chunks_per_row; c += 256) {
    int lr = c / chunks_per_row;
    int kc = (c - lr * chunks_per_row) * 8;
    int gate = lr >> 4, nloc = lr & 15;
    const bf16_t* src = W + (size_t)(gate * H_ + n0 + nloc) * KENC + kc;
    *reinterpret_cast<v8bf*>(lds_w + (size_t)lr * LDW + kc) =
        *reinterpret_cast<const v8bf*>(src);
  }
}

// ---------------------------------------------------------------------------
// Prep kernels
// ---------------------------------------------------------------------------
__global__ void concat_w_kernel(bf16_t* dst, const float* Wa, const float* Wb) {
  int i = blockIdx.x * 256 + threadIdx.x;
  if (i >= G4 * KENC) return;
  int r = i / KENC, k = i - r * KENC;
  float v = (k < E_) ? Wa[(size_t)r * E_ + k] : Wb[(size_t)r * H_ + (k - E_)];
  dst[i] = (bf16_t)v;
}
__global__ void cvt_bf_kernel(bf16_t* dst, const float* src, int n) {
  int i = blockIdx.x * 256 + threadIdx.x;
  if (i < n) dst[i] = (bf16_t)src[i];
}
__global__ void embed_enc_kernel(bf16_t* dst, const float* emb, const int* sent) {
  int i = blockIdx.x * 256 + threadIdx.x;           // [S][B][E]
  if (i >= S_ * B_ * E_) return;
  int e = i & (E_ - 1);
  int sb = i >> 8;                                   // E_ = 256
  int b = sb & (B_ - 1), s = sb >> 6;                // B_ = 64
  int tok = sent[s * B_ + b];
  dst[i] = (bf16_t)emb[(size_t)tok * E_ + e];
}
__global__ void embed_dec_kernel(bf16_t* dst, const float* emb, const int* sent) {
  int i = blockIdx.x * 256 + threadIdx.x;           // [B][S][E]
  if (i >= B_ * S_ * E_) return;
  int e = i & (E_ - 1);
  int bs = i >> 8;
  int s = bs & (S_ - 1), b = bs >> 9;                // S_ = 512
  int tok = sent[s * B_ + b];
  dst[i] = (bf16_t)emb[(size_t)tok * E_ + e];
}
__global__ void init_state_kernel(Ws w, const float* dec_in0) {
  int i = blockIdx.x * 256 + threadIdx.x;
  if (i < B_ * H_) {
    w.Cbuf[i] = 0.f;
    w.Mask[i] = 1.f;                                 // B*S == B*H == 32768
    w.Hbuf[i] = (bf16_t)0.f;
    w.Hbuf[B_ * H_ + i] = (bf16_t)0.f;
  }
  if (i < B_ * E_) w.Decin[i] = (bf16_t)dec_in0[i & (E_ - 1)];
  if (i < 8) w.Sync[i] = 0u;
}

// ---------------------------------------------------------------------------
// Encoder: persistent kernel, 32 WGs x 256 threads, 512 sequential steps.
// WG owns 16 hidden units (n0..n0+15) across all four gates; its 96KB weight
// slice lives in LDS for the whole recurrence.
// ---------------------------------------------------------------------------
__global__ void __launch_bounds__(256)
enc_kernel(Ws w, const float* b_ih, const float* b_hh) {
  extern __shared__ char dynsm[];
  bf16_t* lds_w = reinterpret_cast<bf16_t*>(dynsm);       // [64][LDW]
  __shared__ float gbuf[4][64][16];
  const int tid = threadIdx.x, lane = tid & 31, wave = tid >> 5;
  const int n0 = blockIdx.x * 16;
  const int gate = (2 * wave) >> 2;
  const int rt0 = (2 * wave) & 3;
  unsigned target = 0;

  stage_gate_slice(lds_w, w.Wenc, n0, tid);
  __syncthreads();

  for (int t = 0; t < S_; ++t) {
    const bf16_t* xt  = w.Xenc + (size_t)t * B_ * E_;
    const bf16_t* hin = w.Hbuf + (size_t)(t & 1) * B_ * H_;
    bf16_t* hout      = w.Hbuf + (size_t)((t + 1) & 1) * B_ * H_;

    v8f acc0 = {}; v8f acc1 = {};
    for (int kk = 0; kk < KENC / 32; ++kk) {
      const int k0 = kk * 32;
      v16bf bfrag = load_b16(lds_w, LDW, gate * 16, k0, lane);
      const bf16_t* ab; int ald, kl;
      if (k0 < E_) { ab = xt;  ald = E_; kl = k0; }
      else         { ab = hin; ald = H_; kl = k0 - E_; }
      v16bf a0 = load_a16(ab, ald, rt0 * 16,      kl, lane);
      v16bf a1 = load_a16(ab, ald, rt0 * 16 + 16, kl, lane);
      acc0 = wmma_bf16(a0, bfrag, acc0);
      acc1 = wmma_bf16(a1, bfrag, acc1);
    }
    const int nl = lane & 15, mo = (lane >> 4) << 3;
#pragma unroll
    for (int j = 0; j < 8; ++j) {
      gbuf[gate][rt0 * 16 + mo + j][nl]      = acc0[j];
      gbuf[gate][rt0 * 16 + 16 + mo + j][nl] = acc1[j];
    }
    __syncthreads();

    // LSTM cell update for our 16 hidden units x 64 batches
    for (int u = tid; u < B_ * 16; u += 256) {
      int m = u >> 4, ln = u & 15, n = n0 + ln;
      float gi = gbuf[0][m][ln] + b_ih[n]          + b_hh[n];
      float gf = gbuf[1][m][ln] + b_ih[H_ + n]     + b_hh[H_ + n];
      float gg = gbuf[2][m][ln] + b_ih[2 * H_ + n] + b_hh[2 * H_ + n];
      float go = gbuf[3][m][ln] + b_ih[3 * H_ + n] + b_hh[3 * H_ + n];
      float c  = w.Cbuf[m * H_ + n];
      float cn = sigmoidf_(gf) * c + sigmoidf_(gi) * tanhf(gg);
      float hn = sigmoidf_(go) * tanhf(cn);
      w.Cbuf[m * H_ + n] = cn;
      bf16_t hb = (bf16_t)hn;
      hout[m * H_ + n] = hb;
      w.EncO[((size_t)t * B_ + m) * H_ + n] = hb;
    }
    target += gridDim.x;
    grid_sync(w.Sync, target);
  }
}

// ---------------------------------------------------------------------------
// Context projection: ctxT[b][s][g] = EncO[(s,b),:] . W_ctx[g,:] + b_ctx[g]
// M = 32768, N = 512, K = 512  (bf16 WMMA, 64x64 macro tile per WG)
// ---------------------------------------------------------------------------
__global__ void __launch_bounds__(256)
ctx_kernel(Ws w, const float* b_ctx) {
  const int tid = threadIdx.x, lane = tid & 31, wave = tid >> 5;
  const int rowM = (blockIdx.x >> 3) * 64;
  const int colM = (blockIdx.x & 7) * 64;
#pragma unroll
  for (int ti = 0; ti < 2; ++ti) {
    const int tt = wave * 2 + ti;
    const int row0 = rowM + (tt & 3) * 16;
    const int col0 = colM + (tt >> 2) * 16;
    v8f acc = {};
    for (int kk = 0; kk < H_ / 32; ++kk) {
      v16bf a = load_a16(w.EncO, H_, row0, kk * 32, lane);
      v16bf b = load_b16(w.Wctx, H_, col0, kk * 32, lane);
      acc = wmma_bf16(a, b, acc);
    }
    const int g = col0 + (lane & 15);
    const float bc = b_ctx[g];
#pragma unroll
    for (int j = 0; j < 8; ++j) {
      int r = row0 + ((lane >> 4) << 3) + j;   // r = s*64 + b
      int s = r >> 6, b = r & 63;
      w.Ctx[((size_t)b * S_ + s) * H_ + g] = acc[j] + bc;
    }
  }
}

// ---------------------------------------------------------------------------
// Decoder: persistent kernel, 32 WGs x 256 threads, 32 steps x 4 phases.
// Wdec slice + W_in slice staged in LDS; W_out streamed from L2 (1x/step).
// ---------------------------------------------------------------------------
__global__ void __launch_bounds__(256)
dec_kernel(Ws w, const float* bx, const float* bh, const float* b_in,
           const float* b_out, const float* Vv, float* alphas, int* ptrs) {
  extern __shared__ char dynsm[];
  bf16_t* lds_wdec = reinterpret_cast<bf16_t*>(dynsm);            // [64][LDW]
  bf16_t* lds_win  = lds_wdec + (size_t)64 * LDW;                 // [16][LDWIN]
  __shared__ float gbuf[4][64][16];
  __shared__ float s_inp[H_], s_V[H_], s_alpha[S_];
  __shared__ float s_red[256];
  __shared__ int   s_redi[256];
  __shared__ int   s_idx;
  const int tid = threadIdx.x, lane = tid & 31, wave = tid >> 5;
  const int n0 = blockIdx.x * 16;
  unsigned target = 0;

  stage_gate_slice(lds_wdec, w.Wdec, n0, tid);
  // W_in slice: rows n0..n0+15 of [512][512]
  for (int c = tid; c < 16 * (H_ / 8); c += 256) {
    int lr = c / (H_ / 8);
    int kc = (c - lr * (H_ / 8)) * 8;
    *reinterpret_cast<v8bf*>(lds_win + (size_t)lr * LDWIN + kc) =
        *reinterpret_cast<const v8bf*>(w.Win + (size_t)(n0 + lr) * H_ + kc);
  }
  __syncthreads();

  for (int t = 0; t < T_; ++t) {
    // ---- phase 1: LSTM cell: gates = [Decin | Hbuf] @ Wdec^T + bx + bh ----
    {
      const int gate = (2 * wave) >> 2;
      const int rt0 = (2 * wave) & 3;
      v8f acc0 = {}; v8f acc1 = {};
      for (int kk = 0; kk < KENC / 32; ++kk) {
        const int k0 = kk * 32;
        v16bf bfrag = load_b16(lds_wdec, LDW, gate * 16, k0, lane);
        const bf16_t* ab; int ald, kl;
        if (k0 < E_) { ab = w.Decin; ald = E_; kl = k0; }
        else         { ab = w.Hbuf;  ald = H_; kl = k0 - E_; }
        v16bf a0 = load_a16(ab, ald, rt0 * 16,      kl, lane);
        v16bf a1 = load_a16(ab, ald, rt0 * 16 + 16, kl, lane);
        acc0 = wmma_bf16(a0, bfrag, acc0);
        acc1 = wmma_bf16(a1, bfrag, acc1);
      }
      const int nl = lane & 15, mo = (lane >> 4) << 3;
#pragma unroll
      for (int j = 0; j < 8; ++j) {
        gbuf[gate][rt0 * 16 + mo + j][nl]      = acc0[j];
        gbuf[gate][rt0 * 16 + 16 + mo + j][nl] = acc1[j];
      }
      __syncthreads();
      for (int u = tid; u < B_ * 16; u += 256) {
        int m = u >> 4, ln = u & 15, n = n0 + ln;
        float gi = gbuf[0][m][ln] + bx[n]          + bh[n];
        float gf = gbuf[1][m][ln] + bx[H_ + n]     + bh[H_ + n];
        float gg = gbuf[2][m][ln] + bx[2 * H_ + n] + bh[2 * H_ + n];
        float go = gbuf[3][m][ln] + bx[3 * H_ + n] + bh[3 * H_ + n];
        float c  = w.Cbuf[m * H_ + n];
        float cn = sigmoidf_(gf) * c + sigmoidf_(gi) * tanhf(gg);
        float hn = sigmoidf_(go) * tanhf(cn);
        w.Cbuf[m * H_ + n] = cn;
        w.Ht[m * H_ + n] = (bf16_t)hn;
      }
      target += gridDim.x; grid_sync(w.Sync + 1, target);
    }
    // ---- phase 2: inp = Ht @ W_in^T + b_in  (M=64,N=512,K=512) ----
    {
      const int col0 = blockIdx.x * 16;
      if (wave < 4) {
        const int row0 = wave * 16;
        v8f acc = {};
        for (int kk = 0; kk < H_ / 32; ++kk) {
          v16bf a = load_a16(w.Ht, H_, row0, kk * 32, lane);
          v16bf b = load_b16(lds_win, LDWIN, 0, kk * 32, lane);
          acc = wmma_bf16(a, b, acc);
        }
        const int n = col0 + (lane & 15);
        const float bb = b_in[n];
#pragma unroll
        for (int j = 0; j < 8; ++j) {
          int m = row0 + ((lane >> 4) << 3) + j;
          w.Inp[m * H_ + n] = acc[j] + bb;
        }
      }
      target += gridDim.x; grid_sync(w.Sync + 1, target);
    }
    // ---- phase 3: attention / softmax / argmax / hidden_state (2 batches) --
    for (int bb2 = 0; bb2 < 2; ++bb2) {
      const int b = blockIdx.x * 2 + bb2;
      s_V[tid] = Vv[tid];             s_V[tid + 256] = Vv[tid + 256];
      s_inp[tid] = w.Inp[b * H_ + tid];
      s_inp[tid + 256] = w.Inp[b * H_ + tid + 256];
      __syncthreads();
      float attv[2], mkv[2];
#pragma unroll
      for (int r = 0; r < 2; ++r) {
        const int s = tid + r * 256;
        const float* crow = w.Ctx + ((size_t)b * S_ + s) * H_;
        float a = 0.f;
        for (int g = 0; g < H_; ++g) a += s_V[g] * tanhf(s_inp[g] + crow[g]);
        mkv[r] = w.Mask[b * S_ + s];
        attv[r] = a;
      }
      float mloc = -INFINITY;
#pragma unroll
      for (int r = 0; r < 2; ++r) if (mkv[r] != 0.f) mloc = fmaxf(mloc, attv[r]);
      s_red[tid] = mloc; __syncthreads();
      for (int st = 128; st > 0; st >>= 1) {
        if (tid < st) s_red[tid] = fmaxf(s_red[tid], s_red[tid + st]);
        __syncthreads();
      }
      const float mx = s_red[0]; __syncthreads();
      float ev[2]; float sloc = 0.f;
#pragma unroll
      for (int r = 0; r < 2; ++r) {
        ev[r] = (mkv[r] != 0.f) ? __expf(attv[r] - mx) : 0.f;
        sloc += ev[r];
      }
      s_red[tid] = sloc; __syncthreads();
      for (int st = 128; st > 0; st >>= 1) {
        if (tid < st) s_red[tid] += s_red[tid + st];
        __syncthreads();
      }
      const float inv = 1.f / s_red[0]; __syncthreads();
      float bv = -1.f; int bi = 0x7fffffff;
#pragma unroll
      for (int r = 0; r < 2; ++r) {
        const int s = tid + r * 256;
        const float al = ev[r] * inv;
        s_alpha[s] = al;
        alphas[((size_t)b * T_ + t) * S_ + s] = al;
        if (al > bv || (al == bv && s < bi)) { bv = al; bi = s; }
      }
      s_red[tid] = bv; s_redi[tid] = bi; __syncthreads();
      for (int st = 128; st > 0; st >>= 1) {
        if (tid < st) {
          float v2 = s_red[tid + st]; int i2 = s_redi[tid + st];
          if (v2 > s_red[tid] || (v2 == s_red[tid] && i2 < s_redi[tid])) {
            s_red[tid] = v2; s_redi[tid] = i2;
          }
        }
        __syncthreads();
      }
      if (tid == 0) s_idx = s_redi[0];
      __syncthreads();
      const int idx = s_idx;
#pragma unroll
      for (int r = 0; r < 2; ++r) {
        const int g = tid + r * 256;
        float hs = 0.f;
        for (int s = 0; s < S_; ++s)
          hs += w.Ctx[((size_t)b * S_ + s) * H_ + g] * s_alpha[s];
        w.Hid[b * H_ + g] = (bf16_t)hs;
      }
      if (tid == 0) {
        ptrs[b * T_ + t] = idx;
        w.Mask[b * S_ + idx] = 0.f;
      }
      w.Decin[b * E_ + tid] = w.Demb[((size_t)b * S_ + idx) * E_ + tid];
      __syncthreads();
    }
    target += gridDim.x; grid_sync(w.Sync + 1, target);
    // ---- phase 4: h_new = tanh([Hid | Ht] @ W_out^T + b_out) ----
    {
      const int col0 = blockIdx.x * 16;
      if (wave < 4) {
        const int row0 = wave * 16;
        v8f acc = {};
        for (int kk = 0; kk < (2 * H_) / 32; ++kk) {
          const int k0 = kk * 32;
          const bf16_t* ab; int kl;
          if (k0 < H_) { ab = w.Hid; kl = k0; }
          else         { ab = w.Ht;  kl = k0 - H_; }
          v16bf a = load_a16(ab, H_, row0, kl, lane);
          v16bf b = load_b16(w.Wout, 2 * H_, col0, k0, lane);
          acc = wmma_bf16(a, b, acc);
        }
        const int n = col0 + (lane & 15);
        const float bb = b_out[n];
#pragma unroll
        for (int j = 0; j < 8; ++j) {
          int m = row0 + ((lane >> 4) << 3) + j;
          w.Hbuf[m * H_ + n] = (bf16_t)tanhf(acc[j] + bb);
        }
      }
      target += gridDim.x; grid_sync(w.Sync + 1, target);
    }
  }
}

// ---------------------------------------------------------------------------
extern "C" void kernel_launch(void* const* d_in, const int* in_sizes, int n_in,
                              void* d_out, int out_size, void* d_ws,
                              size_t ws_size, hipStream_t stream) {
  (void)in_sizes; (void)n_in; (void)out_size; (void)ws_size;
  const int*   sent    = (const int*)d_in[0];
  const float* emb_enc = (const float*)d_in[1];
  const float* emb_dec = (const float*)d_in[2];
  const float* W_ih    = (const float*)d_in[3];
  const float* W_hh    = (const float*)d_in[4];
  const float* b_ih    = (const float*)d_in[5];
  const float* b_hh    = (const float*)d_in[6];
  const float* dec_in0 = (const float*)d_in[7];
  const float* Wx      = (const float*)d_in[8];
  const float* bx      = (const float*)d_in[9];
  const float* Wh      = (const float*)d_in[10];
  const float* bh      = (const float*)d_in[11];
  const float* W_out   = (const float*)d_in[12];
  const float* b_out   = (const float*)d_in[13];
  const float* W_in    = (const float*)d_in[14];
  const float* b_in    = (const float*)d_in[15];
  const float* W_ctx   = (const float*)d_in[16];
  const float* b_ctx   = (const float*)d_in[17];
  const float* Vv      = (const float*)d_in[18];

  char* p = (char*)d_ws;
  auto alloc = [&](size_t bytes) -> char* {
    char* r = p; p += (bytes + 255) & ~(size_t)255; return r;
  };
  Ws w;
  w.Wenc  = (bf16_t*)alloc((size_t)G4 * KENC * 2);
  w.Wdec  = (bf16_t*)alloc((size_t)G4 * KENC * 2);
  w.Win   = (bf16_t*)alloc((size_t)H_ * H_ * 2);
  w.Wout  = (bf16_t*)alloc((size_t)H_ * 2 * H_ * 2);
  w.Wctx  = (bf16_t*)alloc((size_t)H_ * H_ * 2);
  w.Xenc  = (bf16_t*)alloc((size_t)S_ * B_ * E_ * 2);
  w.Demb  = (bf16_t*)alloc((size_t)B_ * S_ * E_ * 2);
  w.EncO  = (bf16_t*)alloc((size_t)S_ * B_ * H_ * 2);
  w.Hbuf  = (bf16_t*)alloc((size_t)2 * B_ * H_ * 2);
  w.Ht    = (bf16_t*)alloc((size_t)B_ * H_ * 2);
  w.Hid   = (bf16_t*)alloc((size_t)B_ * H_ * 2);
  w.Decin = (bf16_t*)alloc((size_t)B_ * E_ * 2);
  w.Ctx   = (float*)alloc((size_t)B_ * S_ * H_ * 4);
  w.Cbuf  = (float*)alloc((size_t)B_ * H_ * 4);
  w.Inp   = (float*)alloc((size_t)B_ * H_ * 4);
  w.Mask  = (float*)alloc((size_t)B_ * S_ * 4);
  w.Sync  = (unsigned*)alloc(256);

  float* alphas = (float*)d_out;
  int*   ptrs   = (int*)((float*)d_out + (size_t)B_ * T_ * S_);

  // prep: weight conversion / concat, embedding gathers, state init
  concat_w_kernel<<<(G4 * KENC + 255) / 256, 256, 0, stream>>>(w.Wenc, W_ih, W_hh);
  concat_w_kernel<<<(G4 * KENC + 255) / 256, 256, 0, stream>>>(w.Wdec, Wx, Wh);
  cvt_bf_kernel<<<(H_ * H_ + 255) / 256, 256, 0, stream>>>(w.Win, W_in, H_ * H_);
  cvt_bf_kernel<<<(H_ * 2 * H_ + 255) / 256, 256, 0, stream>>>(w.Wout, W_out, H_ * 2 * H_);
  cvt_bf_kernel<<<(H_ * H_ + 255) / 256, 256, 0, stream>>>(w.Wctx, W_ctx, H_ * H_);
  embed_enc_kernel<<<(S_ * B_ * E_ + 255) / 256, 256, 0, stream>>>(w.Xenc, emb_enc, sent);
  embed_dec_kernel<<<(B_ * S_ * E_ + 255) / 256, 256, 0, stream>>>(w.Demb, emb_dec, sent);
  init_state_kernel<<<(B_ * H_ + 255) / 256, 256, 0, stream>>>(w, dec_in0);

  // persistent encoder recurrence (512 steps, grid-synced, weights in LDS)
  enc_kernel<<<32, 256, ENC_LDS, stream>>>(w, b_ih, b_hh);
  // context projection GEMM
  ctx_kernel<<<(S_ * B_ / 64) * (H_ / 64), 256, 0, stream>>>(w, b_ctx);
  // persistent decoder (32 steps x 4 phases, Wdec/W_in in LDS)
  dec_kernel<<<32, 256, DEC_LDS, stream>>>(w, bx, bh, b_in, b_out, Vv, alphas, ptrs);
}
